// RWKV5_AttentionSubLayer_18159121727562
// MI455X (gfx1250) — matrix operand order repaired and checked by reading, not compiled, not run on hardware
//
#include <hip/hip_runtime.h>
#include <hip/hip_bf16.h>

// ---------------------------------------------------------------------------
// RWKV5 attention sublayer for MI455X (gfx1250, wave32, WMMA bf16 16x16x32,
// async global->LDS staging).  B=4, TT=4096, C=1024, H=16, K=64, chunk T=128.
// ---------------------------------------------------------------------------

#define Bsz   4
#define TTlen 4096
#define Cdim  1024
#define Hn    16
#define Kh    64
#define Tch   128
#define Nch   (TTlen / Tch)          // 32
#define Mrows (Bsz * TTlen)          // 16384

typedef __attribute__((ext_vector_type(16))) __bf16 v16bf;
typedef __attribute__((ext_vector_type(8)))  __bf16 bf16x8;
typedef __attribute__((ext_vector_type(8)))  float  v8f;
typedef __attribute__((ext_vector_type(4)))  float  f32x4;

// async-to-LDS builtin operand types: pointer to GCC-style int4 vector in
// AS1 (global) / AS3 (LDS), per the clang signature
// (int __vector(4) __device__*, int __vector(4) __shared__*, Ii, Ii)
typedef int v4i_ __attribute__((vector_size(16)));
typedef __attribute__((address_space(1))) v4i_ as1_v4i;
typedef __attribute__((address_space(3))) v4i_ as3_v4i;

#if defined(__has_builtin)
#if __has_builtin(__builtin_amdgcn_global_load_async_to_lds_b128) && \
    __has_builtin(__builtin_amdgcn_s_wait_asynccnt)
#define HAVE_ASYNC_LDS 1
#endif
#endif
#ifndef HAVE_ASYNC_LDS
#define HAVE_ASYNC_LDS 0
#endif

// per-lane async 16B copy global -> LDS (ASYNCcnt-tracked, no VGPR round-trip)
// generic->AS1: identical 64-bit value; generic->AS3: LDS offset is addr[31:0]
// so integer casts are exact.
__device__ __forceinline__ void async_cp16(const __bf16* g, __bf16* l) {
#if HAVE_ASYNC_LDS
    __builtin_amdgcn_global_load_async_to_lds_b128(
        (as1_v4i*)(unsigned long long)(size_t)g,
        (as3_v4i*)(unsigned int)(size_t)l, 0, 0);
#else
    *(bf16x8*)l = *(const bf16x8*)g;
#endif
}
__device__ __forceinline__ void async_wait() {
#if HAVE_ASYNC_LDS
    __builtin_amdgcn_s_wait_asynccnt(0);
#endif
}

// ---------------------------------------------------------------------------
// WMMA helpers (bf16 A/B, f32 accumulate). Layouts per CDNA5 ISA 7.12.2:
//   A 16x32: lane l holds row m=l&15, k-base=(l<16)?0:8; two 16B runs.
//   B 32x16 from N-major source src[n][k]: lane col n=l&15, k-base=(l<16)?0:8;
//            two 16B runs (same k order as A).
//   C/D 16x16 f32: element e -> row = e + ((l>=16)?8:0), col = l&15.
// ---------------------------------------------------------------------------
__device__ __forceinline__ v16bf load_a_frag(const __bf16* base, int ld) {
    const int l  = threadIdx.x & 31;
    const int m  = l & 15;
    const int kb = (l >> 4) << 3;
    const __bf16* p = base + m * ld + kb;
    union { v16bf v; bf16x8 h[2]; } u;
    u.h[0] = *(const bf16x8*)(p);
    u.h[1] = *(const bf16x8*)(p + 16);
    return u.v;
}

__device__ __forceinline__ v16bf load_bT_frag(const __bf16* base, int ld) {
    const int l  = threadIdx.x & 31;
    const int n  = l & 15;
    const int kb = (l >> 4) << 3;
    const __bf16* p = base + n * ld + kb;
    union { v16bf v; bf16x8 h[2]; } u;
    u.h[0] = *(const bf16x8*)(p);
    u.h[1] = *(const bf16x8*)(p + 16);
    return u.v;
}

__device__ __forceinline__ v8f wmma_bf16(v16bf a, v16bf b, v8f c) {
    return __builtin_amdgcn_wmma_f32_16x16x32_bf16(
        false, a, false, b, (short)0, c, false, false);
}

// ---------------------------------------------------------------------------
// Elementwise prep kernels (8 elems / thread)
// ---------------------------------------------------------------------------
__global__ __launch_bounds__(256) void cvt_f32_bf16(const float* __restrict__ src,
                                                    __bf16* __restrict__ dst) {
    size_t i = ((size_t)blockIdx.x * 256 + threadIdx.x) * 8;
    f32x4 a = *(const f32x4*)(src + i);
    f32x4 b = *(const f32x4*)(src + i + 4);
    bf16x8 o;
    #pragma unroll
    for (int e = 0; e < 4; ++e) { o[e] = (__bf16)a[e]; o[e + 4] = (__bf16)b[e]; }
    *(bf16x8*)(dst + i) = o;
}

// time-shift mixing: xM = x*tm + x_prev*(1-tm) for r,k,v,g -> bf16
__global__ __launch_bounds__(256) void mix_kernel(
    const float* __restrict__ xq,
    const float* __restrict__ tmr, const float* __restrict__ tmk,
    const float* __restrict__ tmv, const float* __restrict__ tmg,
    __bf16* __restrict__ xr, __bf16* __restrict__ xk,
    __bf16* __restrict__ xv, __bf16* __restrict__ xg) {
    size_t i = ((size_t)blockIdx.x * 256 + threadIdx.x) * 8;   // < B*TT*C
    int c = (int)(i & (Cdim - 1));
    size_t t = (i >> 10) & (TTlen - 1);
    float x[8], xx[8];
    *(f32x4*)&x[0] = *(const f32x4*)(xq + i);
    *(f32x4*)&x[4] = *(const f32x4*)(xq + i + 4);
    if (t == 0) {
        #pragma unroll
        for (int e = 0; e < 8; ++e) xx[e] = 0.f;
    } else {
        *(f32x4*)&xx[0] = *(const f32x4*)(xq + i - Cdim);
        *(f32x4*)&xx[4] = *(const f32x4*)(xq + i - Cdim + 4);
    }
    bf16x8 or_, ok_, ov_, og_;
    #pragma unroll
    for (int e = 0; e < 8; ++e) {
        float mr = tmr[c + e], mk = tmk[c + e], mv = tmv[c + e], mg = tmg[c + e];
        or_[e] = (__bf16)(x[e] * mr + xx[e] * (1.f - mr));
        ok_[e] = (__bf16)(x[e] * mk + xx[e] * (1.f - mk));
        ov_[e] = (__bf16)(x[e] * mv + xx[e] * (1.f - mv));
        og_[e] = (__bf16)(x[e] * mg + xx[e] * (1.f - mg));
    }
    *(bf16x8*)(xr + i) = or_;
    *(bf16x8*)(xk + i) = ok_;
    *(bf16x8*)(xv + i) = ov_;
    *(bf16x8*)(xg + i) = og_;
}

// ---------------------------------------------------------------------------
// NT GEMM: out[m,n] = sum_k A[m,k] * W[n,k].  Block tile 128x64, 8 waves,
// W strip staged in LDS (64x128 K-blocks) via async b128 copies.
// MODE 0: bf16 head-split (B,H,TT,Kh); MODE 1: bf16 SiLU; MODE 2: f32
// ---------------------------------------------------------------------------
template <int MODE>
__global__ __launch_bounds__(256) void gemm_nt(const __bf16* __restrict__ A,
                                               const __bf16* __restrict__ W,
                                               void* __restrict__ out) {
    __shared__ __bf16 wS[64 * 128];
    const int m0  = blockIdx.x * 128;
    const int n0  = blockIdx.y * 64;
    const int tid = threadIdx.x;
    const int w   = tid >> 5;
    const int l   = tid & 31;
    const int mrow = m0 + w * 16;

    v8f acc[4] = {};

    for (int k0 = 0; k0 < Cdim; k0 += 128) {
        __syncthreads();
        // 64x128 bf16 strip = 1024 x 16B; 4 async b128 per thread
        #pragma unroll
        for (int q = 0; q < 4; ++q) {
            int i  = tid * 4 + q;
            int nn = i >> 4, kk = (i & 15) << 3;
            async_cp16(W + (size_t)(n0 + nn) * Cdim + k0 + kk,
                       wS + nn * 128 + kk);
        }
        async_wait();
        __syncthreads();
        if (k0 + 128 < Cdim)
            __builtin_prefetch(A + (size_t)mrow * Cdim + k0 + 128, 0, 0);
        #pragma unroll
        for (int ks = 0; ks < 128; ks += 32) {
            v16bf a = load_a_frag(A + (size_t)mrow * Cdim + k0 + ks, Cdim);
            #pragma unroll
            for (int nt = 0; nt < 4; ++nt) {
                v16bf b = load_bT_frag(wS + (nt * 16) * 128 + ks, 128);
                acc[nt] = wmma_bf16(a, b, acc[nt]);
            }
        }
    }

    const int rb  = (l >> 4) << 3;
    const int col = l & 15;
    #pragma unroll
    for (int nt = 0; nt < 4; ++nt) {
        #pragma unroll
        for (int e = 0; e < 8; ++e) {
            int mm = mrow + rb + e;
            int nn = n0 + nt * 16 + col;
            float vv = acc[nt][e];
            if (MODE == 2) {
                ((float*)out)[(size_t)mm * Cdim + nn] = vv;
            } else if (MODE == 1) {
                float s = vv / (1.f + __expf(-vv));   // SiLU
                ((__bf16*)out)[(size_t)mm * Cdim + nn] = (__bf16)s;
            } else {
                int b = mm >> 12, t = mm & (TTlen - 1);
                int h = nn >> 6,  kk = nn & (Kh - 1);
                ((__bf16*)out)[(((size_t)b * Hn + h) * TTlen + t) * Kh + kk] = (__bf16)vv;
            }
        }
    }
}

// ---------------------------------------------------------------------------
// Chunked RWKV5 attention: one block per (b,h), 8 waves, sequential chunks.
// All four einsums on v_wmma_f32_16x16x32_bf16; f32 state in LDS.
// k chunk staged row-major via async b128; v/kw transposed via bf16x8 loads.
// ---------------------------------------------------------------------------
__global__ __launch_bounds__(256) void rwkv_attn(
    const __bf16* __restrict__ r, const __bf16* __restrict__ k,
    const __bf16* __restrict__ v,
    const float* __restrict__ time_decay, const float* __restrict__ time_faaaa,
    float* __restrict__ xo) {
    __shared__ __bf16 kS[Tch * Kh];        // k row-major  (Tch x Kh)    16 KB
    __shared__ __bf16 vT[Kh * Tch];        // v transposed (Kh x Tch)    16 KB
    __shared__ __bf16 kwT[Kh * Tch];       // (k*wk)^T     (Kh x Tch)    16 KB
    __shared__ __bf16 attS[Tch * Tch];     // decayed att  (Tch x Tch)   32 KB
    __shared__ __bf16 sbT[Kh * Kh];        // state^T bf16 (Kh x Kh)      8 KB
    __shared__ float  sF[Kh * Kh];         // state f32                  16 KB
    __shared__ float  wbTab[Tch];          // decay^i

    const int bh = blockIdx.x;             // 0..63
    const int b = bh >> 4, h = bh & 15;
    const int tid = threadIdx.x;
    const int w = tid >> 5, l = tid & 31;

    const float decay = __expf(-__expf(time_decay[h]));
    const float u     = time_faaaa[h];
    const float wsch  = __powf(decay, (float)Tch);

    if (tid < Tch) wbTab[tid] = __powf(decay, (float)tid);
    for (int i = tid; i < Kh * Kh; i += 256) { sF[i] = 0.f; sbT[i] = (__bf16)0.f; }
    __syncthreads();

    const __bf16* rB = r + (size_t)bh * TTlen * Kh;
    const __bf16* kB = k + (size_t)bh * TTlen * Kh;
    const __bf16* vB = v + (size_t)bh * TTlen * Kh;

    const int rb  = (l >> 4) << 3;
    const int col = l & 15;

    for (int n = 0; n < Nch; ++n) {
        const int t0 = n * Tch;

        // async-stage k chunk row-major (1024 x 16B)
        #pragma unroll
        for (int q = 0; q < 4; ++q) {
            int i  = tid * 4 + q;
            int tt = i >> 3, cc = (i & 7) << 3;
            async_cp16(kB + (size_t)(t0 + tt) * Kh + cc, kS + tt * Kh + cc);
        }
        // stage v^T and (k*wk)^T : bf16x8 global reads, transposed ds stores
        for (int i = tid; i < (Tch * Kh) / 8; i += 256) {
            int t = i >> 3, c0 = (i & 7) << 3;
            bf16x8 v8 = *(const bf16x8*)(vB + (size_t)(t0 + t) * Kh + c0);
            bf16x8 k8 = *(const bf16x8*)(kB + (size_t)(t0 + t) * Kh + c0);
            float wkt = wbTab[(Tch - 1) - t];
            #pragma unroll
            for (int e = 0; e < 8; ++e) {
                vT[(c0 + e) * Tch + t]  = v8[e];
                kwT[(c0 + e) * Tch + t] = (__bf16)((float)k8[e] * wkt);
            }
        }
        async_wait();
        __syncthreads();

        // r fragments for this wave's 16-row strip (used by att and r@s)
        const __bf16* rRow = rB + (size_t)(t0 + w * 16) * Kh;
        v16bf aF0 = load_a_frag(rRow + 0, Kh);
        v16bf aF1 = load_a_frag(rRow + 32, Kh);

        // att = (r @ k^T) ⊙ w_mat   (wave w owns rows [16w, 16w+16))
        for (int jt = 0; jt < 8; ++jt) {
            v8f acc = {};
            acc = wmma_bf16(aF0, load_bT_frag(kS + (jt * 16) * Kh + 0, Kh), acc);
            acc = wmma_bf16(aF1, load_bT_frag(kS + (jt * 16) * Kh + 32, Kh), acc);
            #pragma unroll
            for (int e = 0; e < 8; ++e) {
                int i = w * 16 + rb + e;
                int j = jt * 16 + col;
                float wf = (i > j) ? wbTab[i - j - 1] : ((i == j) ? u : 0.f);
                attS[i * Tch + j] = (__bf16)(acc[e] * wf);
            }
        }
        __syncthreads();

        // out = att @ v + (r @ s) * decay^i    -> xo (B,TT,H,Kh)
        for (int nt = 0; nt < 4; ++nt) {
            v8f acc = {};
            const __bf16* aRow = attS + (w * 16) * Tch;
            #pragma unroll
            for (int kk = 0; kk < Tch; kk += 32) {
                v16bf a  = load_a_frag(aRow + kk, Tch);
                v16bf bb = load_bT_frag(vT + (nt * 16) * Tch + kk, Tch);
                acc = wmma_bf16(a, bb, acc);
            }
            v8f racc = {};
            racc = wmma_bf16(aF0, load_bT_frag(sbT + (nt * 16) * Kh + 0, Kh), racc);
            racc = wmma_bf16(aF1, load_bT_frag(sbT + (nt * 16) * Kh + 32, Kh), racc);
            #pragma unroll
            for (int e = 0; e < 8; ++e) {
                int i = w * 16 + rb + e;
                int c = nt * 16 + col;
                float o = acc[e] + racc[e] * wbTab[i];
                xo[(((size_t)b * TTlen + t0 + i) * Hn + h) * Kh + c] = o;
            }
        }
        __syncthreads();

        // state update: s = ws*s + (k*wk)^T @ v   (16 tiles over 8 waves)
        #pragma unroll
        for (int tt = 0; tt < 2; ++tt) {
            int tile = w * 2 + tt;
            int rt = tile >> 2, ct = tile & 3;
            v8f acc = {};
            #pragma unroll
            for (int kk = 0; kk < Tch; kk += 32) {
                v16bf a  = load_a_frag(kwT + (rt * 16) * Tch + kk, Tch);
                v16bf bb = load_bT_frag(vT + (ct * 16) * Tch + kk, Tch);
                acc = wmma_bf16(a, bb, acc);
            }
            #pragma unroll
            for (int e = 0; e < 8; ++e) {
                int m = rt * 16 + rb + e;      // k-feature
                int c = ct * 16 + col;         // v-feature
                float sn = wsch * sF[m * Kh + c] + acc[e];
                sF[m * Kh + c] = sn;
                sbT[c * Kh + m] = (__bf16)sn;  // transposed bf16 for next r@s
            }
        }
        __syncthreads();
    }
}

// ---------------------------------------------------------------------------
// GroupNorm (H groups of Kh) on xo/8, gated by silu(g) -> bf16 yg
// One wave32 per (b,t,h) group; 2 channels per lane.
// ---------------------------------------------------------------------------
__global__ __launch_bounds__(256) void groupnorm_gate(
    const float* __restrict__ xo, const float* __restrict__ lnw,
    const float* __restrict__ lnb, const __bf16* __restrict__ g,
    __bf16* __restrict__ yg) {
    int grp = blockIdx.x * 8 + (threadIdx.x >> 5);  // 0 .. B*TT*H-1
    int l = threadIdx.x & 31;
    int h = grp & (Hn - 1);
    size_t bt = (size_t)grp >> 4;                   // b*TT + t
    const float* xp = xo + (bt * Hn + h) * Kh;
    float v0 = xp[l] * 0.125f;                      // / head_size_divisor
    float v1 = xp[l + 32] * 0.125f;
    float s  = v0 + v1;
    float s2 = v0 * v0 + v1 * v1;
    #pragma unroll
    for (int m = 16; m; m >>= 1) {
        s  += __shfl_xor(s, m, 32);
        s2 += __shfl_xor(s2, m, 32);
    }
    float mu  = s * (1.f / Kh);
    float var = s2 * (1.f / Kh) - mu * mu;
    float inv = rsqrtf(var + 1e-5f);
    #pragma unroll
    for (int q = 0; q < 2; ++q) {
        int kk = l + q * 32;
        float val = (q == 0 ? v0 : v1);
        float y = (val - mu) * inv * lnw[h * Kh + kk] + lnb[h * Kh + kk];
        float gg = (float)g[bt * Cdim + h * Kh + kk];
        yg[bt * Cdim + h * Kh + kk] = (__bf16)(y * gg);
    }
}

// ---------------------------------------------------------------------------
// Host-side orchestration
// ---------------------------------------------------------------------------
extern "C" void kernel_launch(void* const* d_in, const int* in_sizes, int n_in,
                              void* d_out, int out_size, void* d_ws, size_t ws_size,
                              hipStream_t stream) {
    (void)in_sizes; (void)n_in; (void)out_size; (void)ws_size;
    const float* xq   = (const float*)d_in[0];
    const float* tmk  = (const float*)d_in[1];
    const float* tmv  = (const float*)d_in[2];
    const float* tmr  = (const float*)d_in[3];
    const float* tmg  = (const float*)d_in[4];
    const float* tdec = (const float*)d_in[5];
    const float* tfaa = (const float*)d_in[6];
    const float* Wr   = (const float*)d_in[7];
    const float* Wk   = (const float*)d_in[8];
    const float* Wv   = (const float*)d_in[9];
    const float* Wg   = (const float*)d_in[10];
    const float* Wo   = (const float*)d_in[11];
    const float* lnw  = (const float*)d_in[12];
    const float* lnb  = (const float*)d_in[13];
    float* out = (float*)d_out;

    char* ws = (char*)d_ws;
    const size_t MB = 1024ull * 1024ull;
    const size_t WSZ = (size_t)Cdim * Cdim;                 // 1M elems
    const size_t XSZ = (size_t)Mrows * Cdim;                // 16M elems
    __bf16* wrB = (__bf16*)(ws + 0 * MB);
    __bf16* wkB = (__bf16*)(ws + 2 * MB);
    __bf16* wvB = (__bf16*)(ws + 4 * MB);
    __bf16* wgB = (__bf16*)(ws + 6 * MB);
    __bf16* woB = (__bf16*)(ws + 8 * MB);
    __bf16* xr  = (__bf16*)(ws + 10 * MB);
    __bf16* xk  = (__bf16*)(ws + 42 * MB);
    __bf16* xv  = (__bf16*)(ws + 74 * MB);
    __bf16* xg  = (__bf16*)(ws + 106 * MB);
    __bf16* rH  = (__bf16*)(ws + 138 * MB);   // (B,H,TT,Kh)
    __bf16* kH  = (__bf16*)(ws + 170 * MB);
    __bf16* vH  = (__bf16*)(ws + 202 * MB);
    __bf16* gS  = (__bf16*)(ws + 234 * MB);   // silu(g), row-major
    float*  xo  = (float*)(ws + 10 * MB);     // reuses xr+xk (dead after GEMMs)
    __bf16* yg  = (__bf16*)(ws + 74 * MB);    // reuses xv (dead after GEMMs)

    const unsigned cvtGrid = (unsigned)(WSZ / (256 * 8));
    cvt_f32_bf16<<<cvtGrid, 256, 0, stream>>>(Wr, wrB);
    cvt_f32_bf16<<<cvtGrid, 256, 0, stream>>>(Wk, wkB);
    cvt_f32_bf16<<<cvtGrid, 256, 0, stream>>>(Wv, wvB);
    cvt_f32_bf16<<<cvtGrid, 256, 0, stream>>>(Wg, wgB);
    cvt_f32_bf16<<<cvtGrid, 256, 0, stream>>>(Wo, woB);

    mix_kernel<<<(unsigned)(XSZ / (256 * 8)), 256, 0, stream>>>(
        xq, tmr, tmk, tmv, tmg, xr, xk, xv, xg);

    dim3 gemmGrid(Mrows / 128, Cdim / 64);
    gemm_nt<0><<<gemmGrid, 256, 0, stream>>>(xr, wrB, rH);
    gemm_nt<0><<<gemmGrid, 256, 0, stream>>>(xk, wkB, kH);
    gemm_nt<0><<<gemmGrid, 256, 0, stream>>>(xv, wvB, vH);
    gemm_nt<1><<<gemmGrid, 256, 0, stream>>>(xg, wgB, gS);

    rwkv_attn<<<Bsz * Hn, 256, 0, stream>>>(rH, kH, vH, tdec, tfaa, xo);

    groupnorm_gate<<<(Bsz * TTlen * Hn) / 8, 256, 0, stream>>>(xo, lnw, lnb, gS, yg);

    gemm_nt<2><<<gemmGrid, 256, 0, stream>>>(yg, woB, out);
}
